// RelativeBias_609885356095
// MI455X (gfx1250) — matrix operand order repaired and checked by reading, not compiled, not run on hardware
//
#include <hip/hip_runtime.h>

typedef __attribute__((ext_vector_type(16))) _Float16 v16h;
typedef __attribute__((ext_vector_type(8)))  float    v8f;

#define NPTS 1024
#define NBAT 2
#define NHEAD 16

// Each wave32 computes one 16(j) x 16(head) output tile for a fixed i:
//   D = A(16x64 f16 pos-features) x B(64x16 f16 kernel) + bias
// realized as two v_wmma_f32_16x16x32_f16 (K=0..31 = sin block, K=32..63 = cos block).
__global__ __launch_bounds__(256) void RelativeBias_kernel(
    const float* __restrict__ vec,   // (B, N, 3)
    const float* __restrict__ kern,  // (64, 16)
    const float* __restrict__ bias,  // (16,)
    float* __restrict__ out)         // (B, N, N, 16)
{
    const int wave = threadIdx.x >> 5;   // 0..7 : j-subtile within block
    const int lane = threadIdx.x & 31;
    const int m    = lane & 15;          // row (j) / col (head) index within tile
    const int half = lane >> 4;          // lane-group selector for K striping

    // Block decomposition: grid.x = NBAT * NPTS * (NPTS/128)
    const int jblk = blockIdx.x & 7;                 // NPTS/128 = 8
    const int i    = (blockIdx.x >> 3) & (NPTS - 1);
    const int b    = blockIdx.x >> 13;               // /(8*1024)
    const int j0   = jblk * 128 + wave * 16;
    const int j    = j0 + m;

    // ---- angle for this lane's (i, j) pair (uniform i -> scalar loads) ----
    const float* vi = vec + ((long long)b * NPTS + i) * 3;
    const float* vj = vec + ((long long)b * NPTS + j) * 3;
    const float ax = vi[0], ay = vi[1], az = vi[2];
    const float bx = vj[0], by = vj[1], bz = vj[2];
    const float ra = rsqrtf(ax * ax + ay * ay + az * az);
    const float rb = rsqrtf(bx * bx + by * by + bz * bz);
    float dot = (ax * bx + ay * by + az * bz) * ra * rb;
    dot = fminf(1.0f, fmaxf(-1.0f, dot));
    const float angle = acosf(dot) * 200.0f;         // / 0.005

    // ---- A fragments: lane owns K = 16*half + idx ----
    // freqs[f] = (1/PERIOD) * f/31 = 10*f/31 ; features f=K (sin) and f=32+K (cos)
    v16h a_sin, a_cos;
#pragma unroll
    for (int idx = 0; idx < 16; ++idx) {
        const int k = 16 * half + idx;   // K in [0,32)
        const float ph = angle * (10.0f / 31.0f) * (float)k;
        a_sin[idx] = (_Float16)__sinf(ph);
        a_cos[idx] = (_Float16)__cosf(ph);
    }

    // ---- B fragments from the 64x16 dense kernel (resident in WGP$/L2) ----
    // 32x16 f16 B as packed here (same k order as A): lane L holds N = L&15, K = 16*half + idx
    v16h b_sin, b_cos;
#pragma unroll
    for (int idx = 0; idx < 16; ++idx) {
        const int k = 16 * half + idx;                 // K in [0,32)
        b_sin[idx] = (_Float16)kern[k * NHEAD + m];          // sin rows 0..31
        b_cos[idx] = (_Float16)kern[(k + 32) * NHEAD + m];   // cos rows 32..63
    }

    // ---- accumulator pre-loaded with bias (C/D: N = lane&15 per lane) ----
    const float bv = bias[m];
    v8f c = {bv, bv, bv, bv, bv, bv, bv, bv};

    c = __builtin_amdgcn_wmma_f32_16x16x32_f16(false, a_sin, false, b_sin,
                                               (short)0, c, false, false);
    c = __builtin_amdgcn_wmma_f32_16x16x32_f16(false, a_cos, false, b_cos,
                                               (short)0, c, false, false);

    // ---- stage D tile to LDS, then 2x fully-coalesced b128 stores/lane ----
    // D layout: lane holds column h = m, rows M = r + 8*half  (r = VGPR 0..7)
    __shared__ float tile[8][16][20];   // pad 20: 16B-aligned rows, conflict-free
#pragma unroll
    for (int r = 0; r < 8; ++r)
        tile[wave][r + 8 * half][m] = c[r];

    float* outp = out + (((long long)b * NPTS + i) * NPTS + j0) * NHEAD;

    // wave writes 256 contiguous floats: element e = row*16 + head
    const int e0 = lane * 4;          // first 512B chunk
    const int e1 = 128 + lane * 4;    // second 512B chunk
    const float4 o0 = *(const float4*)&tile[wave][e0 >> 4][e0 & 15];
    const float4 o1 = *(const float4*)&tile[wave][e1 >> 4][e1 & 15];
    *(float4*)(outp + e0) = o0;
    *(float4*)(outp + e1) = o1;
}

extern "C" void kernel_launch(void* const* d_in, const int* in_sizes, int n_in,
                              void* d_out, int out_size, void* d_ws, size_t ws_size,
                              hipStream_t stream) {
    const float* vec  = (const float*)d_in[0];   // (2, 1024, 3)
    const float* kern = (const float*)d_in[1];   // (64, 16)
    const float* bias = (const float*)d_in[2];   // (16,)
    float* out = (float*)d_out;                  // (2, 1024, 1024, 16)

    const int grid = NBAT * NPTS * (NPTS / 128); // 16384 blocks of 8 waves
    RelativeBias_kernel<<<grid, 256, 0, stream>>>(vec, kern, bias, out);
}
